// GPT2Block_91182155694202
// MI455X (gfx1250) — matrix operand order repaired and checked by reading, not compiled, not run on hardware
//
#include <hip/hip_runtime.h>
#include <stdint.h>

// ---------------------------------------------------------------------------
// CDNA5 / gfx1250: wave32, WMMA 16x16x32 f16 -> f32, double-buffered
// async global->LDS tile pipeline (ASYNCcnt in-order completion).
// ---------------------------------------------------------------------------

typedef _Float16 v16h __attribute__((ext_vector_type(16)));
typedef float    v8f  __attribute__((ext_vector_type(8)));

union F16Frag { v16h v; uint32_t u[8]; };
union PackH2  { _Float16 h[2]; uint32_t u; };

static __device__ __forceinline__ v8f wmma_f16(const F16Frag& a, const F16Frag& b, v8f c) {
  return __builtin_amdgcn_wmma_f32_16x16x32_f16(false, a.v, false, b.v, (short)0, c, false, false);
}

static __device__ __forceinline__ v8f zero8() {
  v8f z = {0.f, 0.f, 0.f, 0.f, 0.f, 0.f, 0.f, 0.f};
  return z;
}

// async 16-byte global -> LDS copy (per-lane), tracked by ASYNCcnt
static __device__ __forceinline__ void async_g2l_b128(uint32_t lds_byte_off,
                                                      const void* gaddr) {
  asm volatile("global_load_async_to_lds_b128 %0, %1, off"
               :: "v"(lds_byte_off), "v"(gaddr) : "memory");
}
static __device__ __forceinline__ void wait_async0() {
  asm volatile("s_wait_asynccnt 0x0" ::: "memory");
}
// wait until only the most recent 8 async ops remain outstanding
static __device__ __forceinline__ void wait_async8() {
  asm volatile("s_wait_asynccnt 0x8" ::: "memory");
}

#define N_EMBED 1024
#define T_SEQ   256
#define B_BATCH 64
#define N_HEAD  16
#define HEAD_D  64
#define M_ROWS  (B_BATCH * T_SEQ)   // 16384

// ---------------------------------------------------------------------------
// fp32 [K][N] -> f16 [N][K] transpose+convert (weights), 32x32 LDS tiles
// ---------------------------------------------------------------------------
__global__ __launch_bounds__(256)
void cvt_transpose_f16(const float* __restrict__ src, _Float16* __restrict__ dst,
                       int K, int N) {
  __shared__ float tile[32][33];
  const int kb = blockIdx.y * 32;
  const int nb = blockIdx.x * 32;
  const int tx = threadIdx.x;   // 0..31
  const int ty = threadIdx.y;   // 0..7
#pragma unroll
  for (int i = 0; i < 32; i += 8)
    tile[ty + i][tx] = src[(size_t)(kb + ty + i) * N + nb + tx];
  __syncthreads();
#pragma unroll
  for (int i = 0; i < 32; i += 8)
    dst[(size_t)(nb + ty + i) * K + kb + tx] = (_Float16)tile[tx][ty + i];
}

// ---------------------------------------------------------------------------
// LayerNorm over C=1024, one block (256 threads) per row, f16 output
// ---------------------------------------------------------------------------
__global__ __launch_bounds__(256)
void layernorm_f16(const float* __restrict__ x, const float* __restrict__ w,
                   const float* __restrict__ bia, _Float16* __restrict__ out) {
  const int row = blockIdx.x;
  const int tid = threadIdx.x;
  const float* xr = x + (size_t)row * N_EMBED;
  float v[4], s = 0.f, s2 = 0.f;
#pragma unroll
  for (int i = 0; i < 4; i++) {
    v[i] = xr[tid + 256 * i];
    s += v[i];
    s2 += v[i] * v[i];
  }
  __shared__ float ssum[256], ssq[256];
  ssum[tid] = s; ssq[tid] = s2;
  __syncthreads();
  for (int off = 128; off > 0; off >>= 1) {
    if (tid < off) { ssum[tid] += ssum[tid + off]; ssq[tid] += ssq[tid + off]; }
    __syncthreads();
  }
  const float mean = ssum[0] * (1.0f / N_EMBED);
  const float var  = ssq[0] * (1.0f / N_EMBED) - mean * mean;
  const float rstd = rsqrtf(var + 1e-5f);
  _Float16* orow = out + (size_t)row * N_EMBED;
#pragma unroll
  for (int i = 0; i < 4; i++) {
    const int c = tid + 256 * i;
    orow[c] = (_Float16)((v[i] - mean) * rstd * w[c] + bia[c]);
  }
}

// ---------------------------------------------------------------------------
// WMMA GEMM: out[M,N] = A[M,K](f16) * Wt[N,K](f16)^T + bias, epilogues:
//   MODE 0: store f16           (qkv)
//   MODE 1: GELU, store f16     (fc)
//   MODE 2: +resid, store f32   (proj / out)
// 256 threads (8 waves, 2x4), 128x128 macro tile, K-step 64,
// 2-stage LDS pipeline fed by async global->LDS b128 copies.
// ---------------------------------------------------------------------------
template <int MODE>
__global__ __launch_bounds__(256)
void gemm_wmma(const _Float16* __restrict__ A, const _Float16* __restrict__ Wt,
               const float* __restrict__ bias, const float* __restrict__ resid,
               void* __restrict__ out, int Ndim, int Kdim) {
  constexpr int STR  = 72;                  // halves; 144B row stride (16B-aligned)
  constexpr int TILE = 128 * STR;           // halves per matrix per stage
  __shared__ _Float16 smem[4 * TILE];       // [stage][A | B]; assumed LDS base 0

  const int tid  = threadIdx.x;
  const int lane = tid & 31;
  const int wv   = tid >> 5;
  const int wm   = (wv >> 2) * 64;   // 0 / 64
  const int wn   = (wv & 3) * 32;    // 0 / 32 / 64 / 96
  const int sub  = lane >> 4;
  const int l16  = lane & 15;
  const int bm   = blockIdx.y * 128;
  const int bn   = blockIdx.x * 128;

  // per-thread copy slots: 8 x b128 per tile pair (4 for A, 4 for B)
  int r_[4], c_[4];
#pragma unroll
  for (int i = 0; i < 4; i++) {
    const int seg = tid + i * 256;          // 0..1023
    r_[i] = seg >> 3;                       // row 0..127
    c_[i] = (seg & 7) * 8;                  // half offset in row
  }

  const _Float16* Abase = A  + (size_t)bm * Kdim;
  const _Float16* Bbase = Wt + (size_t)bn * Kdim;

  auto issue_tile = [&](int k0, int stage) {
#pragma unroll
    for (int i = 0; i < 4; i++) {
      const uint32_t la = (uint32_t)((stage * 2 * TILE + r_[i] * STR + c_[i]) * 2);
      async_g2l_b128(la,            Abase + (size_t)r_[i] * Kdim + k0 + c_[i]);
      async_g2l_b128(la + TILE * 2, Bbase + (size_t)r_[i] * Kdim + k0 + c_[i]);
    }
  };

  v8f acc[4][2];
#pragma unroll
  for (int i = 0; i < 4; i++)
#pragma unroll
    for (int j = 0; j < 2; j++) acc[i][j] = zero8();

  const int nsteps = Kdim >> 6;             // K-step 64
  issue_tile(0, 0);

  for (int it = 0; it < nsteps; it++) {
    if (it + 1 < nsteps) {
      issue_tile((it + 1) << 6, (it + 1) & 1);
      wait_async8();                        // in-order: current tile's 8 are done
    } else {
      wait_async0();
    }
    __syncthreads();

    const _Float16* lA = smem + (it & 1) * 2 * TILE;
    const _Float16* lB = lA + TILE;

#pragma unroll
    for (int ks = 0; ks < 2; ks++) {
      const int kbase = ks * 32;
      F16Frag bf[2];
#pragma unroll
      for (int ct = 0; ct < 2; ct++)
#pragma unroll
        for (int v = 0; v < 8; v++) {
          const int kk = kbase + sub * 16 + v * 2;
          bf[ct].u[v] = *(const uint32_t*)&lB[(wn + ct * 16 + l16) * STR + kk];
        }
#pragma unroll
      for (int rt = 0; rt < 4; rt++) {
        F16Frag af;
#pragma unroll
        for (int v = 0; v < 8; v++) {
          const int kk = kbase + sub * 16 + v * 2;
          af.u[v] = *(const uint32_t*)&lA[(wm + rt * 16 + l16) * STR + kk];
        }
        acc[rt][0] = wmma_f16(af, bf[0], acc[rt][0]);
        acc[rt][1] = wmma_f16(af, bf[1], acc[rt][1]);
      }
    }
    __syncthreads();   // protect this stage before it is refilled at it+2
  }

  // epilogue (C/D layout: n = l16, m = 8*sub + r)
#pragma unroll
  for (int rt = 0; rt < 4; rt++)
#pragma unroll
    for (int ct = 0; ct < 2; ct++) {
      const int n = bn + wn + ct * 16 + l16;
      const float bv = bias[n];
#pragma unroll
      for (int r = 0; r < 8; r++) {
        const int m = bm + wm + rt * 16 + sub * 8 + r;
        const size_t off = (size_t)m * Ndim + n;
        float val = acc[rt][ct][r] + bv;
        if constexpr (MODE == 0) {
          ((_Float16*)out)[off] = (_Float16)val;
        } else if constexpr (MODE == 1) {
          val = 0.5f * val * (1.0f + erff(val * 0.70710678118f));
          ((_Float16*)out)[off] = (_Float16)val;
        } else {
          ((float*)out)[off] = resid[off] + val;
        }
      }
    }
}

// ---------------------------------------------------------------------------
// Flash attention: one wave per (b, h, 32-row q-tile); key chunks of 32;
// online softmax; S and P*V via WMMA.
// qkv layout: [B,T,3C] f16 (q | k | v each C wide); y: [B,T,C] f16
// ---------------------------------------------------------------------------
__global__ __launch_bounds__(32)
void attention_flash(const _Float16* __restrict__ qkv, _Float16* __restrict__ y) {
  const int lane = threadIdx.x;
  const int sub  = lane >> 4;
  const int l16  = lane & 15;
  const int q0   = blockIdx.x * 32;
  const int bh   = blockIdx.y;
  const int b    = bh >> 4;
  const int h    = bh & 15;
  const size_t rs = 3 * N_EMBED;

  const _Float16* qb = qkv + (size_t)b * T_SEQ * rs + h * HEAD_D;
  const _Float16* kb = qb + N_EMBED;
  const _Float16* vb = qb + 2 * N_EMBED;

  __shared__ _Float16 P[32 * 40];

  F16Frag qf[2][2];
#pragma unroll
  for (int rt = 0; rt < 2; rt++)
#pragma unroll
    for (int c = 0; c < 2; c++)
#pragma unroll
      for (int v = 0; v < 8; v++) {
        const int kd = c * 32 + sub * 16 + v * 2;
        const int t  = q0 + rt * 16 + l16;
        qf[rt][c].u[v] = *(const uint32_t*)(qb + (size_t)t * rs + kd);
      }

  v8f Oacc[2][4];
#pragma unroll
  for (int rt = 0; rt < 2; rt++)
#pragma unroll
    for (int nt = 0; nt < 4; nt++) Oacc[rt][nt] = zero8();

  float mst[2][8], lst[2][8];
#pragma unroll
  for (int rt = 0; rt < 2; rt++)
#pragma unroll
    for (int r = 0; r < 8; r++) { mst[rt][r] = -1e30f; lst[rt][r] = 0.f; }

  const int nch = q0 / 32 + 1;
  for (int ci = 0; ci < nch; ci++) {
    const int k0 = ci * 32;

    F16Frag kf[2][2];
#pragma unroll
    for (int kt = 0; kt < 2; kt++)
#pragma unroll
      for (int c = 0; c < 2; c++)
#pragma unroll
        for (int v = 0; v < 8; v++) {
          const int kd = c * 32 + sub * 16 + v * 2;
          const int t  = k0 + kt * 16 + l16;
          kf[kt][c].u[v] = *(const uint32_t*)(kb + (size_t)t * rs + kd);
        }

    v8f s[2][2];
#pragma unroll
    for (int rt = 0; rt < 2; rt++)
#pragma unroll
      for (int kt = 0; kt < 2; kt++) {
        v8f tmp = wmma_f16(qf[rt][0], kf[kt][0], zero8());
        s[rt][kt] = wmma_f16(qf[rt][1], kf[kt][1], tmp);
      }

    const bool diag = (ci == nch - 1);
#pragma unroll
    for (int rt = 0; rt < 2; rt++) {
#pragma unroll
      for (int r = 0; r < 8; r++) {
        const int row = q0 + rt * 16 + sub * 8 + r;
        float s0 = s[rt][0][r] * 0.125f;
        float s1 = s[rt][1][r] * 0.125f;
        if (diag) {
          if (k0 + l16 > row)      s0 = -1e30f;
          if (k0 + 16 + l16 > row) s1 = -1e30f;
        }
        float mx = fmaxf(s0, s1);
#pragma unroll
        for (int off = 1; off < 16; off <<= 1)
          mx = fmaxf(mx, __shfl_xor(mx, off, 32));
        const float mnew  = fmaxf(mst[rt][r], mx);
        const float alpha = __expf(mst[rt][r] - mnew);
        const float p0 = __expf(s0 - mnew);
        const float p1 = __expf(s1 - mnew);
        float rsum = p0 + p1;
#pragma unroll
        for (int off = 1; off < 16; off <<= 1)
          rsum += __shfl_xor(rsum, off, 32);
        lst[rt][r] = lst[rt][r] * alpha + rsum;
        mst[rt][r] = mnew;
        P[(rt * 16 + sub * 8 + r) * 40 + l16]      = (_Float16)p0;
        P[(rt * 16 + sub * 8 + r) * 40 + 16 + l16] = (_Float16)p1;
#pragma unroll
        for (int nt = 0; nt < 4; nt++) Oacc[rt][nt][r] *= alpha;
      }
    }
    __syncthreads();

    F16Frag pf[2];
#pragma unroll
    for (int rt = 0; rt < 2; rt++)
#pragma unroll
      for (int v = 0; v < 8; v++) {
        const int kk = sub * 16 + v * 2;
        pf[rt].u[v] = *(const uint32_t*)&P[(rt * 16 + l16) * 40 + kk];
      }

#pragma unroll
    for (int nt = 0; nt < 4; nt++) {
      F16Frag vf;
#pragma unroll
      for (int v = 0; v < 8; v++) {
        const int key = k0 + sub * 16 + v * 2;
        PackH2 pk;
        pk.h[0] = vb[(size_t)key * rs + nt * 16 + l16];
        pk.h[1] = vb[(size_t)(key + 1) * rs + nt * 16 + l16];
        vf.u[v] = pk.u;
      }
      Oacc[0][nt] = wmma_f16(pf[0], vf, Oacc[0][nt]);
      Oacc[1][nt] = wmma_f16(pf[1], vf, Oacc[1][nt]);
    }
    __syncthreads();
  }

#pragma unroll
  for (int rt = 0; rt < 2; rt++)
#pragma unroll
    for (int r = 0; r < 8; r++) {
      const float inv = 1.0f / lst[rt][r];
      const int t = q0 + rt * 16 + sub * 8 + r;
      _Float16* yr = y + ((size_t)(b * T_SEQ + t)) * N_EMBED + h * HEAD_D;
#pragma unroll
      for (int nt = 0; nt < 4; nt++)
        yr[nt * 16 + l16] = (_Float16)(Oacc[rt][nt][r] * inv);
    }
}

// ---------------------------------------------------------------------------
// Host launcher
// ---------------------------------------------------------------------------
extern "C" void kernel_launch(void* const* d_in, const int* in_sizes, int n_in,
                              void* d_out, int out_size, void* d_ws, size_t ws_size,
                              hipStream_t stream) {
  const float* x      = (const float*)d_in[0];
  const float* ln1w   = (const float*)d_in[1];
  const float* ln1b   = (const float*)d_in[2];
  const float* w_attn = (const float*)d_in[3];
  const float* b_attn = (const float*)d_in[4];
  const float* w_proj = (const float*)d_in[5];
  const float* b_proj = (const float*)d_in[6];
  const float* ln2w   = (const float*)d_in[7];
  const float* ln2b   = (const float*)d_in[8];
  const float* w_fc   = (const float*)d_in[9];
  const float* b_fc   = (const float*)d_in[10];
  const float* w_out  = (const float*)d_in[11];
  const float* b_out  = (const float*)d_in[12];

  // workspace layout (halves unless noted); weights stored TRANSPOSED [N][K]
  _Float16* wAttnT = (_Float16*)d_ws;                         // [3072][1024]
  _Float16* wProjT = wAttnT + (size_t)N_EMBED * 3 * N_EMBED;  // [1024][1024]
  _Float16* wFcT   = wProjT + (size_t)N_EMBED * N_EMBED;      // [4096][1024]
  _Float16* wOutT  = wFcT   + (size_t)N_EMBED * 4 * N_EMBED;  // [1024][4096]
  _Float16* hln    = wOutT  + (size_t)4 * N_EMBED * N_EMBED;  // 16384*1024
  _Float16* qkvh   = hln    + (size_t)M_ROWS * N_EMBED;       // 16384*4096 (qkv, reused as mlp h)
  _Float16* hf     = qkvh;
  _Float16* yh     = qkvh   + (size_t)M_ROWS * 4 * N_EMBED;   // 16384*1024
  float*    x2     = (float*)(yh + (size_t)M_ROWS * N_EMBED); // 16384*1024 f32

  const dim3 tb(32, 8);
  // 1) weights -> f16, transposed to [N][K]
  cvt_transpose_f16<<<dim3(3 * N_EMBED / 32, N_EMBED / 32), tb, 0, stream>>>(
      w_attn, wAttnT, N_EMBED, 3 * N_EMBED);
  cvt_transpose_f16<<<dim3(N_EMBED / 32, N_EMBED / 32), tb, 0, stream>>>(
      w_proj, wProjT, N_EMBED, N_EMBED);
  cvt_transpose_f16<<<dim3(4 * N_EMBED / 32, N_EMBED / 32), tb, 0, stream>>>(
      w_fc, wFcT, N_EMBED, 4 * N_EMBED);
  cvt_transpose_f16<<<dim3(N_EMBED / 32, 4 * N_EMBED / 32), tb, 0, stream>>>(
      w_out, wOutT, 4 * N_EMBED, N_EMBED);

  // 2) ln1(x) -> hln (f16)
  layernorm_f16<<<M_ROWS, 256, 0, stream>>>(x, ln1w, ln1b, hln);

  // 3) qkv = hln @ w_attn + b_attn  (f16 out)
  gemm_wmma<0><<<dim3(3 * N_EMBED / 128, M_ROWS / 128), 256, 0, stream>>>(
      hln, wAttnT, b_attn, nullptr, qkvh, 3 * N_EMBED, N_EMBED);

  // 4) attention -> yh (f16, heads merged)
  attention_flash<<<dim3(T_SEQ / 32, B_BATCH * N_HEAD), 32, 0, stream>>>(qkvh, yh);

  // 5) x2 = x + yh @ w_proj + b_proj  (f32)
  gemm_wmma<2><<<dim3(N_EMBED / 128, M_ROWS / 128), 256, 0, stream>>>(
      yh, wProjT, b_proj, x, x2, N_EMBED, N_EMBED);

  // 6) ln2(x2) -> hln (f16)
  layernorm_f16<<<M_ROWS, 256, 0, stream>>>(x2, ln2w, ln2b, hln);

  // 7) hf = gelu(hln @ w_fc + b_fc)  (f16)
  gemm_wmma<1><<<dim3(4 * N_EMBED / 128, M_ROWS / 128), 256, 0, stream>>>(
      hln, wFcT, b_fc, nullptr, hf, 4 * N_EMBED, N_EMBED);

  // 8) out = x2 + hf @ w_out + b_out  (f32)
  gemm_wmma<2><<<dim3(N_EMBED / 128, M_ROWS / 128), 256, 0, stream>>>(
      hf, wOutT, b_out, x2, (float*)d_out, N_EMBED, 4 * N_EMBED);
}
